// ProbAttention_82386062672132
// MI455X (gfx1250) — matrix operand order repaired and checked by reading, not compiled, not run on hardware
//
#include <hip/hip_runtime.h>
#include <math.h>

typedef __attribute__((ext_vector_type(16))) _Float16 v16h;
typedef __attribute__((ext_vector_type(8)))  float    v8f;

#define Bc 4
#define Lc 4096
#define Hc 8
#define Dc 64
#define Sc 45
#define Uc 45
#define BHc (Bc*Hc)

// ---------------------------------------------------------------------------
// CDNA5 async global->LDS copy (ASYNCcnt path), 16 bytes per lane.
// ---------------------------------------------------------------------------
__device__ __forceinline__ void async_load16(unsigned ldsOff, const float* gptr) {
    asm volatile("global_load_async_to_lds_b128 %0, %1, off"
                 :
                 : "v"(ldsOff), "v"((unsigned long long)(size_t)gptr)
                 : "memory");
}
__device__ __forceinline__ void wait_async0() {
    asm volatile("s_wait_asynccnt 0" ::: "memory");
}

// ---------------------------------------------------------------------------
// Kernel 1: mean of V over L for each (b,h,d)
// ---------------------------------------------------------------------------
__global__ __launch_bounds__(256) void vmean_kernel(const float* __restrict__ V,
                                                    float* __restrict__ vmean) {
    int bh = blockIdx.x;                 // 0..31
    int b = bh >> 3, h = bh & 7;
    int d = threadIdx.x & 63;
    int part = threadIdx.x >> 6;         // 0..3
    float s = 0.f;
    int l0 = part * (Lc / 4);
    for (int l = l0; l < l0 + (Lc / 4); ++l)
        s += V[(((long)(b * Lc + l)) * Hc + h) * Dc + d];
    __shared__ float red[256];
    red[threadIdx.x] = s;
    __syncthreads();
    if (part == 0)
        vmean[bh * Dc + d] = (red[d] + red[64 + d] + red[128 + d] + red[192 + d]) * (1.0f / Lc);
}

// ---------------------------------------------------------------------------
// Kernel 2: fill output with broadcast mean(V); out layout (B, L, H, D)
// ---------------------------------------------------------------------------
__global__ __launch_bounds__(256) void fill_kernel(const float* __restrict__ vmean,
                                                   float* __restrict__ out) {
    const long total = (long)Bc * Lc * Hc * Dc;
    for (long i = (long)blockIdx.x * blockDim.x + threadIdx.x; i < total;
         i += (long)gridDim.x * blockDim.x) {
        int d = (int)(i & 63);
        int h = (int)((i >> 6) & 7);
        int b = (int)(i >> 21);
        out[i] = vmean[(b * Hc + h) * Dc + d];
    }
}

// ---------------------------------------------------------------------------
// Kernel 3: sampled sparsity measure M[b,h,q] = max_s(qk) - sum_s(qk)/L
// one wave32 per (b,h,q)
// ---------------------------------------------------------------------------
__global__ __launch_bounds__(256) void sampled_score_kernel(const float* __restrict__ Q,
                                                            const float* __restrict__ K,
                                                            const int* __restrict__ samp,
                                                            float* __restrict__ M) {
    __shared__ float qrow[8][Dc];
    int wid = threadIdx.x >> 5;
    int lane = threadIdx.x & 31;
    long gw = (long)blockIdx.x * 8 + wid;     // 0 .. B*H*L-1
    int q  = (int)(gw & (Lc - 1));
    int bh = (int)(gw >> 12);
    int b = bh >> 3, h = bh & 7;

    const float* qp = Q + (((long)(b * Lc + q)) * Hc + h) * Dc;
    qrow[wid][lane]      = qp[lane];
    qrow[wid][lane + 32] = qp[lane + 32];
    __syncthreads();

    float mx = -INFINITY, sm = 0.f;
    for (int s = lane; s < Sc; s += 32) {
        int idx = samp[q * Sc + s];
        const float* kp = K + (((long)(b * Lc + idx)) * Hc + h) * Dc;
        float dot = 0.f;
        #pragma unroll
        for (int d = 0; d < Dc; ++d) dot += qrow[wid][d] * kp[d];
        mx = fmaxf(mx, dot);
        sm += dot;
    }
    #pragma unroll
    for (int off = 16; off > 0; off >>= 1) {
        mx = fmaxf(mx, __shfl_xor(mx, off, 32));
        sm += __shfl_xor(sm, off, 32);
    }
    if (lane == 0)
        M[(long)bh * Lc + q] = mx - sm * (1.0f / Lc);
}

// ---------------------------------------------------------------------------
// Kernel 4: top-45 per (b,h) (rank order, ties -> lower index like jax top_k)
// plus an ascending-sorted copy for the final scatter positions.
// ---------------------------------------------------------------------------
__global__ __launch_bounds__(256) void topk_kernel(const float* __restrict__ M,
                                                   int* __restrict__ topIdx,
                                                   int* __restrict__ sortIdx) {
    __shared__ float vals[Lc];
    __shared__ float rv[256];
    __shared__ int   ri[256];
    int bh = blockIdx.x;
    int tid = threadIdx.x;
    for (int i = tid; i < Lc; i += 256) vals[i] = M[(long)bh * Lc + i];
    __syncthreads();

    for (int it = 0; it < Uc; ++it) {
        float bv = -INFINITY; int bi = 0x7fffffff;
        for (int i = tid; i < Lc; i += 256) {
            float v = vals[i];
            if (v > bv || (v == bv && i < bi)) { bv = v; bi = i; }
        }
        rv[tid] = bv; ri[tid] = bi;
        __syncthreads();
        for (int s2 = 128; s2 > 0; s2 >>= 1) {
            if (tid < s2) {
                float ov = rv[tid + s2]; int oi = ri[tid + s2];
                if (ov > rv[tid] || (ov == rv[tid] && oi < ri[tid])) { rv[tid] = ov; ri[tid] = oi; }
            }
            __syncthreads();
        }
        if (tid == 0) { topIdx[bh * Uc + it] = ri[0]; vals[ri[0]] = -INFINITY; }
        __syncthreads();
    }
    if (tid == 0) {
        int tmp[Uc];
        for (int i = 0; i < Uc; ++i) tmp[i] = topIdx[bh * Uc + i];
        for (int i = 1; i < Uc; ++i) {
            int key = tmp[i]; int j = i - 1;
            while (j >= 0 && tmp[j] > key) { tmp[j + 1] = tmp[j]; --j; }
            tmp[j + 1] = key;
        }
        for (int i = 0; i < Uc; ++i) sortIdx[bh * Uc + i] = tmp[i];
    }
}

// ---------------------------------------------------------------------------
// Kernel 5: flash-attention (WMMA f16 -> f32) for the 45 selected queries.
// One block = one (b,h); 12 waves (384 threads); 32-key tiles.
// K/V tiles are double-buffered in LDS as f32, filled by
// GLOBAL_LOAD_ASYNC_TO_LDS_B128 (ASYNCcnt) so the copy of tile t+1 overlaps
// the WMMA/softmax work on tile t. f32->f16 happens at fragment build.
// O row j (top-k rank order) scatters to row sortIdx[j]  (reference quirk).
// ---------------------------------------------------------------------------
__global__ __launch_bounds__(384) void attn_kernel(const float* __restrict__ Q,
                                                   const float* __restrict__ Kg,
                                                   const float* __restrict__ Vg,
                                                   const int* __restrict__ topIdx,
                                                   const int* __restrict__ sortIdx,
                                                   float* __restrict__ out) {
    __shared__ _Float16 Qr[48 * 64];       // padded 45 -> 48 query rows, f16
    __shared__ float    Kt[2][32 * 64];    // double-buffered f32 tiles
    __shared__ float    Vt[2][32 * 64];
    __shared__ float    Sb[48 * 32];       // score tile
    __shared__ _Float16 Pb[48 * 32];       // exp(S - m) tile
    __shared__ float mrow[48], lrow[48], srow[48];

    int bh = blockIdx.x;
    int b = bh >> 3, h = bh & 7;
    int tid = threadIdx.x;
    int wid = tid >> 5, lane = tid & 31;
    int n16 = lane & 15, hi = lane >> 4;

    // stage Q_red (f16), zero padding rows
    for (int i = tid; i < 48 * 64; i += 384) {
        int j = i >> 6, d = i & 63;
        float v = 0.f;
        if (j < Uc) {
            int ql = topIdx[bh * Uc + j];
            v = Q[(((long)(b * Lc + ql)) * Hc + h) * Dc + d];
        }
        Qr[i] = (_Float16)v;
    }
    if (tid < 48) { mrow[tid] = -INFINITY; lrow[tid] = 0.f; }

    v8f acc = {};                          // O tile for this wave
    int mtO = wid >> 2, ntd = wid & 3;     // 3 x 4 grid of (Mtile, Dtile)
    int mtS = wid >> 1, ntS = wid & 1;     // 3 x 2 grid of score tiles (wid<6)

    // issue async copy of K/V tile `kb` into buffer `sel`
    auto issue_tile = [&](int kb, int sel) {
        // 1024 chunks of 16B: 512 for K, 512 for V (32 rows x 16 chunks)
        for (int i = tid; i < 1024; i += 384) {
            int isV = i >> 9;
            int j = i & 511;
            int r = j >> 4, c4 = (j & 15) << 2;
            const float* gp = (isV ? Vg : Kg) +
                              ((((long)(b * Lc + kb + r)) * Hc + h) * Dc + c4);
            float* lp = (isV ? &Vt[sel][0] : &Kt[sel][0]) + r * 64 + c4;
            async_load16((unsigned)(size_t)lp, gp);
        }
    };

    issue_tile(0, 0);
    __syncthreads();                       // covers Qr/mrow init too

    for (int t = 0; t < Lc / 32; ++t) {
        int cur = t & 1;
        wait_async0();                     // tile t resident in LDS
        __syncthreads();
        if (t + 1 < Lc / 32) issue_tile((t + 1) * 32, cur ^ 1);

        // ---- score tiles: S = Qr (48x64) * K_tile^T (64x32) ----
        if (wid < 6) {
            v8f cs = {};
            #pragma unroll
            for (int ks = 0; ks < 2; ++ks) {
                v16h a, bf;
                #pragma unroll
                for (int v = 0; v < 8; ++v) {
                    int kA = ((v < 4) ? 0 : 16) + hi * 8 + (v & 3) * 2;
                    a[2 * v]     = Qr[(mtS * 16 + n16) * 64 + ks * 32 + kA];
                    a[2 * v + 1] = Qr[(mtS * 16 + n16) * 64 + ks * 32 + kA + 1];
                    int kB = ((v < 4) ? 0 : 16) + hi * 8 + (v & 3) * 2;
                    bf[2 * v]     = (_Float16)Kt[cur][(ntS * 16 + n16) * 64 + ks * 32 + kB];
                    bf[2 * v + 1] = (_Float16)Kt[cur][(ntS * 16 + n16) * 64 + ks * 32 + kB + 1];
                }
                cs = __builtin_amdgcn_wmma_f32_16x16x32_f16(false, a, false, bf,
                                                            (short)0, cs, false, false);
            }
            #pragma unroll
            for (int j = 0; j < 8; ++j)
                Sb[(mtS * 16 + j + hi * 8) * 32 + ntS * 16 + n16] = cs[j];
        }
        __syncthreads();

        // ---- online softmax update, one thread per row ----
        if (tid < 48) {
            float mo = mrow[tid];
            float tm = -INFINITY;
            float sv[32];
            #pragma unroll
            for (int c = 0; c < 32; ++c) { sv[c] = Sb[tid * 32 + c]; tm = fmaxf(tm, sv[c]); }
            float mn = fmaxf(mo, tm);
            float sc = expf(mo - mn);      // 0 on first tile (mo = -inf)
            float ls = 0.f;
            #pragma unroll
            for (int c = 0; c < 32; ++c) {
                float p = expf(sv[c] - mn);
                ls += p;
                Pb[tid * 32 + c] = (_Float16)p;
            }
            mrow[tid] = mn;
            lrow[tid] = lrow[tid] * sc + ls;
            srow[tid] = sc;
        }
        __syncthreads();

        // ---- rescale O and accumulate P (48x32) * V_tile (32x64) ----
        {
            #pragma unroll
            for (int j = 0; j < 8; ++j)
                acc[j] *= srow[mtO * 16 + j + hi * 8];
            v16h a, bf;
            #pragma unroll
            for (int v = 0; v < 8; ++v) {
                int kA = ((v < 4) ? 0 : 16) + hi * 8 + (v & 3) * 2;
                a[2 * v]     = Pb[(mtO * 16 + n16) * 32 + kA];
                a[2 * v + 1] = Pb[(mtO * 16 + n16) * 32 + kA + 1];
                int kB = ((v < 4) ? 0 : 16) + hi * 8 + (v & 3) * 2;
                bf[2 * v]     = (_Float16)Vt[cur][kB * 64 + ntd * 16 + n16];
                bf[2 * v + 1] = (_Float16)Vt[cur][(kB + 1) * 64 + ntd * 16 + n16];
            }
            acc = __builtin_amdgcn_wmma_f32_16x16x32_f16(false, a, false, bf,
                                                         (short)0, acc, false, false);
        }
        __syncthreads();
    }

    // ---- epilogue: normalize and scatter to sorted positions ----
    #pragma unroll
    for (int j = 0; j < 8; ++j) {
        int rn = mtO * 16 + j + hi * 8;
        if (rn < Uc) {
            int gl = sortIdx[bh * Uc + rn];
            out[(((long)(b * Lc + gl)) * Hc + h) * Dc + ntd * 16 + n16] = acc[j] / lrow[rn];
        }
    }
}

// ---------------------------------------------------------------------------
extern "C" void kernel_launch(void* const* d_in, const int* in_sizes, int n_in,
                              void* d_out, int out_size, void* d_ws, size_t ws_size,
                              hipStream_t stream) {
    const float* queries = (const float*)d_in[0];
    const float* keys    = (const float*)d_in[1];
    const float* values  = (const float*)d_in[2];
    const int*   samp    = (const int*)d_in[3];
    float* out = (float*)d_out;

    // workspace layout
    float* Mbuf    = (float*)d_ws;                 // B*H*L = 131072 floats
    float* vmean   = Mbuf + (long)BHc * Lc;        // 2048 floats
    int*   topIdx  = (int*)(vmean + BHc * Dc);     // 32*45 ints
    int*   sortIdx = topIdx + BHc * Uc;            // 32*45 ints

    vmean_kernel<<<BHc, 256, 0, stream>>>(values, vmean);
    fill_kernel<<<4096, 256, 0, stream>>>(vmean, out);
    sampled_score_kernel<<<(BHc * Lc) / 8, 256, 0, stream>>>(queries, keys, samp, Mbuf);
    topk_kernel<<<BHc, 256, 0, stream>>>(Mbuf, topIdx, sortIdx);
    attn_kernel<<<BHc, 384, 0, stream>>>(queries, keys, values, topIdx, sortIdx, out);
}
